// Model_26345329394311
// MI455X (gfx1250) — matrix operand order repaired and checked by reading, not compiled, not run on hardware
//
#include <hip/hip_runtime.h>

// ---------------------------------------------------------------------------
// C51 DQN forward for MI455X (gfx1250, wave32, WMMA).
// - All matmuls (3 convs as implicit im2col, 2 FCs) run on
//   v_wmma_f32_16x16x32_bf16 (bf16 A/B, f32 accumulate).
// - Activations are NHWC and the im2col K axis is (ky, kx, c), so every
//   lane's A-fragment K-runs are memory-contiguous -> b128 loads.
// - Weights pre-swizzled into per-lane WMMA B-fragment order -> 2xb128/tile.
// - Conv weight panels staged block-wide into LDS via the gfx1250 async
//   path (global_load_async_to_lds_b128 + s_wait_asynccnt).
// - A/B fragments hold contiguous k = 16*(lane/16) + e; D uses the wave32 accumulator layout.
// ---------------------------------------------------------------------------

typedef __attribute__((ext_vector_type(16))) __bf16 bf16x16;
typedef __attribute__((ext_vector_type(8)))  float  f32x8;

union Frag16 {
    bf16x16        v;
    unsigned short u[16];
    uint4          q[2];
};

// float -> bf16 bits, round-to-nearest-even
__device__ __forceinline__ unsigned short f2bf(float f) {
    unsigned int u = __builtin_bit_cast(unsigned int, f);
    if ((u & 0x7fffffffu) > 0x7f800000u) {          // NaN: keep quiet
        return (unsigned short)((u >> 16) | 0x0040u);
    }
    unsigned int rnd = 0x7fffu + ((u >> 16) & 1u);
    return (unsigned short)((u + rnd) >> 16);
}

// gfx1250 async DMA: one 16B chunk global -> LDS, tracked by ASYNCcnt
__device__ __forceinline__ void async_b128_to_lds(unsigned lds_off, const void* g) {
    asm volatile("global_load_async_to_lds_b128 %0, %1, off"
                 :: "v"(lds_off), "v"(g) : "memory");
}
__device__ __forceinline__ void wait_asynccnt0() {
    asm volatile("s_wait_asynccnt 0" ::: "memory");
}

// ---------------------------------------------------------------------------
// Conversion kernels (one-time, tiny vs the 116 MB input read)
// ---------------------------------------------------------------------------

// NCHW f32 -> NHWC bf16 with scale (input preprocessing x/256)
__global__ __launch_bounds__(256) void nchw_to_nhwc_bf16(
        const float* __restrict__ src, unsigned short* __restrict__ dst,
        int C, int H, int Wd, int n, float scale) {
    int i = (int)(blockIdx.x * 256 + threadIdx.x);
    if (i >= n) return;
    const int c  = i % C;
    const int t  = i / C;
    const int xq = t % Wd;
    const int t2 = t / Wd;
    const int y  = t2 % H;
    const int b  = t2 / H;
    dst[i] = f2bf(src[(((size_t)b * C + c) * H + y) * Wd + xq] * scale);
}

// Weights -> per-lane WMMA B-fragment order (bf16, zero-pad cols >= NSRC).
// dst[((kt*NT + nt)*32 + lane)*16 + e] = B[kt*32 + (lane>>4)*16 + e][nt*16 + (lane&15)]
// MODE 0: conv OIHW source, GEMM K axis is (ky, kx, c)
// MODE 1: dense [K][NSRC] source (k-major)
// MODE 2: dense [K][NSRC] source, K remapped s*64+c -> c*49+s (NCHW flatten)
template <int K, int NPAD, int NSRC, int MODE, int C = 1, int KH = 1, int KW = 1>
__global__ __launch_bounds__(256) void w_to_frag(
        const float* __restrict__ w, unsigned short* __restrict__ dst) {
    constexpr int NT = NPAD / 16;
    int i = (int)(blockIdx.x * 256 + threadIdx.x);
    if (i >= K * NPAD) return;
    const int e    = i & 15;
    const int lane = (i >> 4) & 31;
    const int tile = i >> 9;
    const int kt   = tile / NT;
    const int nt   = tile - kt * NT;
    const int k    = kt * 32 + (lane >> 4) * 16 + e;
    const int n    = nt * 16 + (lane & 15);
    float v = 0.0f;
    if (n < NSRC) {
        if constexpr (MODE == 0) {
            const int ky = k / (KW * C);
            const int r  = k - ky * (KW * C);
            const int kx = r / C;
            const int c  = r - kx * C;
            v = w[(((size_t)n * C + c) * KH + ky) * KW + kx];
        } else if constexpr (MODE == 1) {
            v = w[(size_t)k * NSRC + n];
        } else {                       // MODE 2: conv3-NHWC -> NCHW flatten remap
            const int s  = k >> 6;     // spatial 0..48
            const int ch = k & 63;     // channel 0..63
            v = w[(size_t)(ch * 49 + s) * NSRC + n];
        }
    }
    dst[i] = f2bf(v);
}

// ---------------------------------------------------------------------------
// Implicit-im2col GEMM over NHWC activations, WMMA bf16 -> f32, fused
// bias + ReLU, NHWC (== row-major [M][N]) bf16 output.
// One wave owns a 16-row block and NTW n-tiles (A fragment reused NTW times).
// LDS_B: block stages the full swizzled B panel via the async DMA path.
// ---------------------------------------------------------------------------
template <int BN, int C, int IH, int IW, int KH, int KW, int S,
          int OH, int OW, int N, int NTW, bool LDS_B>
__global__ __launch_bounds__(256) void conv_gemm_wmma(
        const unsigned short* __restrict__ Ain,   // NHWC bf16 [BN][IH][IW][C]
        const unsigned short* __restrict__ Bw,    // swizzled fragment layout
        const float*          __restrict__ bias,  // [N]
        unsigned short*       __restrict__ Out)   // NHWC bf16 [BN][OH][OW][N]
{
    constexpr int K   = C * KH * KW;
    constexpr int M   = BN * OH * OW;
    constexpr int MT  = M / 16;
    constexpr int NT  = N / 16;
    constexpr int NG  = NT / NTW;                  // n-groups (waves per row block)
    constexpr int OSP = OH * OW;
    constexpr int KWC = KW * C;                    // one kernel-window row
    constexpr int IWC = IW * C;                    // one image row
    static_assert(M % 16 == 0 && N % 16 == 0 && K % 32 == 0, "tiles");
    static_assert(KWC % 8 == 0, "A-fragment runs must not cross ky rows");

    extern __shared__ unsigned short bpanel[];
    if constexpr (LDS_B) {
        constexpr int CHUNKS = (K * N) / 8;        // 16-byte chunks
        const unsigned lbase = (unsigned)(size_t)(void*)bpanel;
        for (int ch = (int)threadIdx.x; ch < CHUNKS; ch += 256) {
            async_b128_to_lds(lbase + (unsigned)ch * 16u, Bw + (size_t)ch * 8);
        }
        wait_asynccnt0();
        __syncthreads();
    }

    const int wid = (int)((blockIdx.x * blockDim.x + threadIdx.x) >> 5);
    if (wid >= MT * NG) return;                    // wave-uniform: EXEC stays full
    const int lane = (int)(threadIdx.x & 31);
    const int half = lane >> 4;
    const int mt   = wid / NG;
    const int ng   = wid - mt * NG;

    // A addressing: this lane owns row (lane&15); NHWC window base
    const int row = mt * 16 + (lane & 15);
    const int b   = row / OSP;
    const int sp  = row - b * OSP;
    const int oy  = sp / OW;
    const int ox  = sp - oy * OW;
    const unsigned short* abase =
        Ain + ((size_t)(b * IH + oy * S) * IW + ox * S) * C;

    const unsigned short* bsrc;
    if constexpr (LDS_B) bsrc = bpanel; else bsrc = Bw;

    f32x8 acc[NTW] = {};
    for (int k0 = 0; k0 < K; k0 += 32) {
        const int kt = k0 >> 5;
        if constexpr (!LDS_B) {
            if (k0 + 32 < K)   // warm next swizzled B row (global_prefetch_b8)
                __builtin_prefetch(Bw + ((size_t)(kt + 1) * NT) * 512, 0, 1);
        }
        Frag16 af;
#pragma unroll
        for (int e = 0; e < 16; ++e) {
            // A layout matches B swizzle: lanes 0-15: K=e; lanes 16-31: K=16+e.
            // K axis = (ky, kx, c): contiguous within a window row -> b128s.
            const int kk = k0 + half * 16 + e;
            const int ky = kk / KWC;
            const int rr = kk - ky * KWC;
            af.u[e] = abase[ky * IWC + rr];
        }
#pragma unroll
        for (int j = 0; j < NTW; ++j) {
            const unsigned short* bptr =
                bsrc + (((size_t)kt * NT + (ng * NTW + j)) * 32 + lane) * 16;
            Frag16 bf;
            bf.q[0] = ((const uint4*)bptr)[0];
            bf.q[1] = ((const uint4*)bptr)[1];
            acc[j] = __builtin_amdgcn_wmma_f32_16x16x32_bf16(
                         false, af.v, false, bf.v, (short)0, acc[j], false, false);
        }
    }

    // D layout: VGPR v -> row v (+8 for lane half 1), col = lane&15.
    // NHWC output == row-major [M][N]: no div/mod in the store path.
#pragma unroll
    for (int j = 0; j < NTW; ++j) {
        const int col = (ng * NTW + j) * 16 + (lane & 15);
        const float bv = bias[col];
#pragma unroll
        for (int v = 0; v < 8; ++v) {
            const int drow = mt * 16 + v + half * 8;
            float val = acc[j][v] + bv;
            val = val > 0.0f ? val : 0.0f;        // ReLU
            Out[(size_t)drow * N + col] = f2bf(val);
        }
    }
}

// ---------------------------------------------------------------------------
// FC2 (256 -> 51, padded 64) fused with softmax, action broadcast, and
// expected value. One wave: 16 batch rows x all 64 cols (4 accumulators).
// ---------------------------------------------------------------------------
__global__ __launch_bounds__(256) void fc2_softmax_kernel(
        const unsigned short* __restrict__ H,    // [1024][256] bf16 (fc1 out)
        const unsigned short* __restrict__ W2,   // swizzled [256][64] bf16
        const float*          __restrict__ fb2,  // [51]
        const float*          __restrict__ z,    // [51]
        float*                __restrict__ out)  // dist[1024*6*51] ++ res[1024*6]
{
    __shared__ float logits[8][16][64];           // 32 KB of 320 KB LDS
    const int wl   = (int)(threadIdx.x >> 5);
    const int wt   = (int)blockIdx.x * 8 + wl;    // 0..63 row tiles
    const int lane = (int)(threadIdx.x & 31);
    const int half = lane >> 4;
    const int mrow = wt * 16 + (lane & 15);
    const int c    = lane & 15;

    f32x8 acc[4] = {};
    for (int k0 = 0; k0 < 256; k0 += 32) {
        const int kt = k0 >> 5;
        Frag16 af;
#pragma unroll
        for (int e = 0; e < 16; ++e) {
            const int kk = k0 + half * 16 + e;
            af.u[e] = H[(size_t)mrow * 256 + kk];
        }
#pragma unroll
        for (int j = 0; j < 4; ++j) {
            const unsigned short* bptr = W2 + (((size_t)kt * 4 + j) * 32 + lane) * 16;
            Frag16 bf;
            bf.q[0] = ((const uint4*)bptr)[0];
            bf.q[1] = ((const uint4*)bptr)[1];
            acc[j] = __builtin_amdgcn_wmma_f32_16x16x32_bf16(
                         false, af.v, false, bf.v, (short)0, acc[j], false, false);
        }
    }

#pragma unroll
    for (int v = 0; v < 8; ++v) {
        const int r = v + half * 8;
        logits[wl][r][c]      = acc[0][v] + (c      < 51 ? fb2[c]      : 0.0f);
        logits[wl][r][c + 16] = acc[1][v] + (c + 16 < 51 ? fb2[c + 16] : 0.0f);
        logits[wl][r][c + 32] = acc[2][v] + (c + 32 < 51 ? fb2[c + 32] : 0.0f);
        logits[wl][r][c + 48] = acc[3][v] + (c + 48 < 51 ? fb2[c + 48] : 0.0f);
    }
    __syncthreads();

    if (lane < 16) {                              // one lane per batch row
        const float* l = logits[wl][lane];
        float mx = l[0];
        for (int i = 1; i < 51; ++i) mx = fmaxf(mx, l[i]);
        float sum = 0.0f;
        for (int i = 0; i < 51; ++i) sum += __expf(l[i] - mx);
        const float inv  = 1.0f / sum;
        const int   bidx = wt * 16 + lane;
        float ev = 0.0f;
        for (int i = 0; i < 51; ++i) {
            const float p = __expf(l[i] - mx) * inv;
            ev += p * z[i];
            float* dp = out + (size_t)bidx * (6 * 51) + i;
#pragma unroll
            for (int a = 0; a < 6; ++a) dp[a * 51] = p;   // identical per action
        }
        float* rp = out + (size_t)1024 * 6 * 51 + (size_t)bidx * 6;
#pragma unroll
        for (int a = 0; a < 6; ++a) rp[a] = ev;
    }
}

// ---------------------------------------------------------------------------
// Host-side orchestration (graph-capture safe: launches only)
// ---------------------------------------------------------------------------
extern "C" void kernel_launch(void* const* d_in, const int* in_sizes, int n_in,
                              void* d_out, int out_size, void* d_ws, size_t ws_size,
                              hipStream_t stream) {
    (void)in_sizes; (void)n_in; (void)out_size; (void)ws_size;

    const float* x   = (const float*)d_in[0];
    const float* cw1 = (const float*)d_in[1];
    const float* cb1 = (const float*)d_in[2];
    const float* cw2 = (const float*)d_in[3];
    const float* cb2 = (const float*)d_in[4];
    const float* cw3 = (const float*)d_in[5];
    const float* cb3 = (const float*)d_in[6];
    const float* fw1 = (const float*)d_in[7];
    const float* fb1 = (const float*)d_in[8];
    const float* fw2 = (const float*)d_in[9];
    const float* fb2 = (const float*)d_in[10];
    const float* z   = (const float*)d_in[11];
    // d_in[12] = n_out (== 6, baked into fc2_softmax_kernel)

    // ---- workspace layout (bf16 elements), ~103 MB total ----
    unsigned short* W = (unsigned short*)d_ws;
    const size_t NXB = 28901376;   // 1024*84*84*4   (NHWC)
    const size_t NA1 = 13107200;   // 1024*20*20*32  (NHWC)
    const size_t NA2 = 5308416;    // 1024*9*9*64    (NHWC)
    const size_t NA3 = 3211264;    // 1024*7*7*64    (NHWC)
    const size_t NH1 = 262144;     // 1024*256
    unsigned short* xb   = W;
    unsigned short* a1   = xb  + NXB;
    unsigned short* a2   = a1  + NA1;
    unsigned short* a3   = a2  + NA2;
    unsigned short* h1   = a3  + NA3;
    unsigned short* w1b  = h1  + NH1;          //  256*32
    unsigned short* w2b  = w1b + 256 * 32;     //  512*64
    unsigned short* w3b  = w2b + 512 * 64;     //  576*64
    unsigned short* fw1b = w3b + 576 * 64;     // 3136*256
    unsigned short* fw2b = fw1b + 3136 * 256;  //  256*64

    // ---- input -> NHWC bf16, weights -> WMMA fragment order ----
    nchw_to_nhwc_bf16<<<(int)(NXB / 256), 256, 0, stream>>>(
        x, xb, 4, 84, 84, (int)NXB, 1.0f / 256.0f);
    w_to_frag< 256,  32,  32, 0,  4, 8, 8><<<  32, 256, 0, stream>>>(cw1, w1b);
    w_to_frag< 512,  64,  64, 0, 32, 4, 4><<< 128, 256, 0, stream>>>(cw2, w2b);
    w_to_frag< 576,  64,  64, 0, 64, 3, 3><<< 144, 256, 0, stream>>>(cw3, w3b);
    w_to_frag<3136, 256, 256, 2          ><<<3136, 256, 0, stream>>>(fw1, fw1b);
    w_to_frag< 256,  64,  51, 1          ><<<  64, 256, 0, stream>>>(fw2, fw2b);

    // ---- layer GEMMs (8 waves/block; wave = 16 rows x NTW n-tiles) ----
    conv_gemm_wmma<1024, 4, 84, 84, 8, 8, 4, 20, 20, 32, 2, true>
        <<<3200, 256, 256 * 32 * 2, stream>>>(xb, w1b, cb1, a1);    // 25600 waves
    conv_gemm_wmma<1024, 32, 20, 20, 4, 4, 2, 9, 9, 64, 4, true>
        <<<648, 256, 512 * 64 * 2, stream>>>(a1, w2b, cb2, a2);     // 5184 waves
    conv_gemm_wmma<1024, 64, 9, 9, 3, 3, 1, 7, 7, 64, 4, true>
        <<<392, 256, 576 * 64 * 2, stream>>>(a2, w3b, cb3, a3);     // 3136 waves
    conv_gemm_wmma<1024, 3136, 1, 1, 1, 1, 1, 1, 1, 256, 4, false>
        <<<32, 256, 0, stream>>>(a3, fw1b, fb1, h1);                // 256 waves

    // ---- FC2 + softmax + broadcast + expected value ----
    fc2_softmax_kernel<<<8, 256, 0, stream>>>(h1, fw2b, fb2, z, (float*)d_out);
}
